// MVAEdecoder_44281112822551
// MI455X (gfx1250) — compile-verified, matrix operand order checked by reading, not run on hardware
//
#include <hip/hip_runtime.h>
#include <stdint.h>

// ---------------------------------------------------------------------------
// MVAE blended-expert decoder for gfx1250 (MI455X).
// Heavy layers (B=4096, E=8): bf16 WMMA 16x16x32, f32 accumulate.
// Compute-bound (~255 GFLOP vs ~200MB traffic) -> matrix pipes, not HBM.
// Wave tile 32M x 64N: 12 b128 loads per 8 WMMAs (fragment-BW optimized).
// ---------------------------------------------------------------------------

typedef __bf16 bfel;
typedef bfel v16bf __attribute__((ext_vector_type(16)));
typedef float v8f  __attribute__((ext_vector_type(8)));
typedef unsigned int v4u __attribute__((ext_vector_type(4)));
typedef unsigned short u16;

union Frag { v4u u[2]; v16bf v; };

__device__ __forceinline__ u16 f2bf(float f) {
  unsigned int u = __float_as_uint(f);
  u += 0x7FFFu + ((u >> 16) & 1u);       // round to nearest even
  return (u16)(u >> 16);
}
__device__ __forceinline__ float bf2f(u16 h) {
  return __uint_as_float(((unsigned int)h) << 16);
}
__device__ __forceinline__ float eluf(float x) {
  return x > 0.0f ? x : (__expf(x) - 1.0f);
}

// dims
static constexpr int Bn  = 4096;
static constexpr int En  = 8;
static constexpr int Hn  = 1024;
static constexpr int XD  = 800;
static constexpr int ZD  = 64;
static constexpr int ODn = 768;
static constexpr int GH  = 128;
static constexpr int K0  = XD + ZD;   // 864
static constexpr int K12 = Hn + ZD;   // 1088

// ---------------------------------------------------------------------------
// Elementwise helpers
// ---------------------------------------------------------------------------
__global__ void k_f32_to_bf16(const float* __restrict__ s, u16* __restrict__ d, int n) {
  int i = blockIdx.x * blockDim.x + threadIdx.x;
  int stride = gridDim.x * blockDim.x;
  for (; i < n; i += stride) d[i] = f2bf(s[i]);
}

// g[b, 0:800] = x[b], g[b, 800:864] = z[b]  (bf16)
__global__ void k_concat_xz(const float* __restrict__ x, const float* __restrict__ z,
                            u16* __restrict__ g) {
  int i = blockIdx.x * blockDim.x + threadIdx.x;
  int stride = gridDim.x * blockDim.x;
  const int n = Bn * K0;
  for (; i < n; i += stride) {
    int b = i / K0, c = i - b * K0;
    float v = (c < XD) ? x[(size_t)b * XD + c] : z[(size_t)b * ZD + (c - XD)];
    g[i] = f2bf(v);
  }
}

// hA[b, 1024:1088] = hB[b, 1024:1088] = z[b]  (bf16)
__global__ void k_zfill(const float* __restrict__ z, u16* __restrict__ hA, u16* __restrict__ hB) {
  int i = blockIdx.x * blockDim.x + threadIdx.x;
  int stride = gridDim.x * blockDim.x;
  const int n = Bn * ZD;
  for (; i < n; i += stride) {
    int b = i / ZD, c = i - b * ZD;
    u16 v = f2bf(z[i]);
    hA[(size_t)b * K12 + Hn + c] = v;
    hB[(size_t)b * K12 + Hn + c] = v;
  }
}

// ---------------------------------------------------------------------------
// Gate GEMM: out[b,o] = elu( sum_k A[b,k]*W[o,k] + bias[o] )  (bf16 in/out)
// One 16x16 tile per wave; 4 waves/block cover 64 M-rows.
// ---------------------------------------------------------------------------
__global__ __launch_bounds__(128) void k_gate_gemm(
    const u16* __restrict__ A, const u16* __restrict__ W,
    const float* __restrict__ bias, u16* __restrict__ out,
    int K, int O)
{
  const int lane = threadIdx.x, wv = threadIdx.y;
  const int m0 = (blockIdx.x * 4 + wv) * 16;
  const int n0 = blockIdx.y * 16;
  const int mrow = lane & 15, hi = lane >> 4;
  const u16* Ap = A + (size_t)(m0 + mrow) * K;
  const u16* Wp = W + (size_t)(n0 + mrow) * K;
  v8f c = {};
  for (int k = 0; k < K; k += 32) {
    const int o1 = k + hi * 8, o2 = o1 + 16;
    Frag a, b;
    a.u[0] = *(const v4u*)(Ap + o1);  a.u[1] = *(const v4u*)(Ap + o2);
    b.u[0] = *(const v4u*)(Wp + o1);  b.u[1] = *(const v4u*)(Wp + o2);
    c = __builtin_amdgcn_wmma_f32_16x16x32_bf16(false, a.v, false, b.v,
                                                (short)0, c, false, false);
  }
  const float bv = bias[n0 + mrow];
#pragma unroll
  for (int r = 0; r < 8; ++r) {
    const int row = m0 + 8 * hi + r;
    out[(size_t)row * O + n0 + mrow] = f2bf(eluf(c[r] + bv));
  }
}

// ---------------------------------------------------------------------------
// Gate layer 3 + softmax -> bc[B,8]  (tiny: VALU)
// ---------------------------------------------------------------------------
__global__ void k_gate3_softmax(const u16* __restrict__ h, const float* __restrict__ Wg3,
                                const float* __restrict__ bg3, float* __restrict__ bc) {
  int b = blockIdx.x * blockDim.x + threadIdx.x;
  if (b >= Bn) return;
  const u16* hr = h + (size_t)b * GH;
  float logits[En];
  float mx = -1e30f;
#pragma unroll
  for (int e = 0; e < En; ++e) {
    float s = bg3[e];
    const float* w = Wg3 + (size_t)e * GH;
    for (int k = 0; k < GH; ++k) s += bf2f(hr[k]) * w[k];
    logits[e] = s;
    mx = fmaxf(mx, s);
  }
  float denom = 0.0f, ex[En];
#pragma unroll
  for (int e = 0; e < En; ++e) { ex[e] = __expf(logits[e] - mx); denom += ex[e]; }
  const float inv = 1.0f / denom;
#pragma unroll
  for (int e = 0; e < En; ++e) bc[(size_t)b * En + e] = ex[e] * inv;
}

// ---------------------------------------------------------------------------
// Blended expert layer:
//   out[b,n] = act( sum_e bc[b,e] * ( sum_k A[b,k]*W[e,n,k] + bias[e,n] ) )
// Wave tile: 32M x 64N (2 A-frags x 4 B-frags -> 8 WMMAs per 12 b128 loads).
// 4 waves/block cover 128 M-rows; identical W rows across waves hit L0.
// ---------------------------------------------------------------------------
template <bool ACT, bool OUTF32>
__global__ __launch_bounds__(128) void k_blended(
    const u16* __restrict__ A, int lda,
    const u16* __restrict__ W,        // [E, O, K] bf16
    const float* __restrict__ bias,   // [E, O]
    const float* __restrict__ bc,     // [B, 8]
    void* __restrict__ outp, int ldo,
    int K, int O)
{
  const int lane = threadIdx.x, wv = threadIdx.y;
  const int m0 = (blockIdx.x * 4 + wv) * 32;
  const int n0 = blockIdx.y * 64;
  const int mrow = lane & 15, hi = lane >> 4;
  const u16* Ap0 = A + (size_t)(m0 + mrow) * lda;
  const u16* Ap1 = A + (size_t)(m0 + 16 + mrow) * lda;
  const float* bcrow0 = bc + (size_t)(m0 + 8 * hi) * En;
  const float* bcrow1 = bc + (size_t)(m0 + 16 + 8 * hi) * En;

  v8f acc00 = {}, acc01 = {}, acc02 = {}, acc03 = {};
  v8f acc10 = {}, acc11 = {}, acc12 = {}, acc13 = {};

  for (int e = 0; e < En; ++e) {
    const u16* Wp0 = W + ((size_t)e * O + n0 +  0 + mrow) * K;
    const u16* Wp1 = W + ((size_t)e * O + n0 + 16 + mrow) * K;
    const u16* Wp2 = W + ((size_t)e * O + n0 + 32 + mrow) * K;
    const u16* Wp3 = W + ((size_t)e * O + n0 + 48 + mrow) * K;
    v8f c00 = {}, c01 = {}, c02 = {}, c03 = {};
    v8f c10 = {}, c11 = {}, c12 = {}, c13 = {};
    for (int k = 0; k < K; k += 32) {
      const int o1 = k + hi * 8, o2 = o1 + 16;
      Frag a0, a1, b0, b1, b2, b3;
      a0.u[0] = *(const v4u*)(Ap0 + o1);  a0.u[1] = *(const v4u*)(Ap0 + o2);
      a1.u[0] = *(const v4u*)(Ap1 + o1);  a1.u[1] = *(const v4u*)(Ap1 + o2);
      b0.u[0] = *(const v4u*)(Wp0 + o1);  b0.u[1] = *(const v4u*)(Wp0 + o2);
      b1.u[0] = *(const v4u*)(Wp1 + o1);  b1.u[1] = *(const v4u*)(Wp1 + o2);
      b2.u[0] = *(const v4u*)(Wp2 + o1);  b2.u[1] = *(const v4u*)(Wp2 + o2);
      b3.u[0] = *(const v4u*)(Wp3 + o1);  b3.u[1] = *(const v4u*)(Wp3 + o2);
      c00 = __builtin_amdgcn_wmma_f32_16x16x32_bf16(false, a0.v, false, b0.v, (short)0, c00, false, false);
      c01 = __builtin_amdgcn_wmma_f32_16x16x32_bf16(false, a0.v, false, b1.v, (short)0, c01, false, false);
      c02 = __builtin_amdgcn_wmma_f32_16x16x32_bf16(false, a0.v, false, b2.v, (short)0, c02, false, false);
      c03 = __builtin_amdgcn_wmma_f32_16x16x32_bf16(false, a0.v, false, b3.v, (short)0, c03, false, false);
      c10 = __builtin_amdgcn_wmma_f32_16x16x32_bf16(false, a1.v, false, b0.v, (short)0, c10, false, false);
      c11 = __builtin_amdgcn_wmma_f32_16x16x32_bf16(false, a1.v, false, b1.v, (short)0, c11, false, false);
      c12 = __builtin_amdgcn_wmma_f32_16x16x32_bf16(false, a1.v, false, b2.v, (short)0, c12, false, false);
      c13 = __builtin_amdgcn_wmma_f32_16x16x32_bf16(false, a1.v, false, b3.v, (short)0, c13, false, false);
    }
    const float bv0 = bias[(size_t)e * O + n0 +  0 + mrow];
    const float bv1 = bias[(size_t)e * O + n0 + 16 + mrow];
    const float bv2 = bias[(size_t)e * O + n0 + 32 + mrow];
    const float bv3 = bias[(size_t)e * O + n0 + 48 + mrow];
#pragma unroll
    for (int r = 0; r < 8; ++r) {
      const float w0 = bcrow0[r * En + e];
      const float w1 = bcrow1[r * En + e];
      acc00[r] += w0 * (c00[r] + bv0);
      acc01[r] += w0 * (c01[r] + bv1);
      acc02[r] += w0 * (c02[r] + bv2);
      acc03[r] += w0 * (c03[r] + bv3);
      acc10[r] += w1 * (c10[r] + bv0);
      acc11[r] += w1 * (c11[r] + bv1);
      acc12[r] += w1 * (c12[r] + bv2);
      acc13[r] += w1 * (c13[r] + bv3);
    }
  }

#pragma unroll
  for (int r = 0; r < 8; ++r) {
    const int row0 = m0 + 8 * hi + r;
    const int row1 = row0 + 16;
    float v00 = acc00[r], v01 = acc01[r], v02 = acc02[r], v03 = acc03[r];
    float v10 = acc10[r], v11 = acc11[r], v12 = acc12[r], v13 = acc13[r];
    if (ACT) {
      v00 = eluf(v00); v01 = eluf(v01); v02 = eluf(v02); v03 = eluf(v03);
      v10 = eluf(v10); v11 = eluf(v11); v12 = eluf(v12); v13 = eluf(v13);
    }
    const size_t b0i = (size_t)row0 * ldo + n0 + mrow;
    const size_t b1i = (size_t)row1 * ldo + n0 + mrow;
    if (OUTF32) {
      float* o = (float*)outp;
      o[b0i] = v00; o[b0i + 16] = v01; o[b0i + 32] = v02; o[b0i + 48] = v03;
      o[b1i] = v10; o[b1i + 16] = v11; o[b1i + 32] = v12; o[b1i + 48] = v13;
    } else {
      u16* o = (u16*)outp;
      o[b0i] = f2bf(v00); o[b0i + 16] = f2bf(v01);
      o[b0i + 32] = f2bf(v02); o[b0i + 48] = f2bf(v03);
      o[b1i] = f2bf(v10); o[b1i + 16] = f2bf(v11);
      o[b1i + 32] = f2bf(v12); o[b1i + 48] = f2bf(v13);
    }
  }
}

// ---------------------------------------------------------------------------
// Launch
// ---------------------------------------------------------------------------
extern "C" void kernel_launch(void* const* d_in, const int* in_sizes, int n_in,
                              void* d_out, int out_size, void* d_ws, size_t ws_size,
                              hipStream_t stream) {
  const float* x   = (const float*)d_in[0];
  const float* z   = (const float*)d_in[1];
  const float* Wg1 = (const float*)d_in[2];
  const float* bg1 = (const float*)d_in[3];
  const float* Wg2 = (const float*)d_in[4];
  const float* bg2 = (const float*)d_in[5];
  const float* Wg3 = (const float*)d_in[6];
  const float* bg3 = (const float*)d_in[7];
  const float* W0  = (const float*)d_in[8];
  const float* b0  = (const float*)d_in[9];
  const float* W1  = (const float*)d_in[10];
  const float* b1  = (const float*)d_in[11];
  const float* W2  = (const float*)d_in[12];
  const float* b2  = (const float*)d_in[13];
  const float* W3  = (const float*)d_in[14];
  const float* b3  = (const float*)d_in[15];
  float* out = (float*)d_out;

  // workspace carve-up (256B aligned); ~55 MB total
  char* p = (char*)d_ws;
  auto carve = [&](size_t bytes) -> char* {
    char* r = p; p += (bytes + 255) & ~(size_t)255; return r;
  };
  u16*   gbuf = (u16*)  carve((size_t)Bn * K0  * 2);   // [B,864]  bf16
  u16*   wg1b = (u16*)  carve((size_t)GH * K0  * 2);   // [128,864]
  u16*   wg2b = (u16*)  carve((size_t)GH * GH  * 2);   // [128,128]
  u16*   g1   = (u16*)  carve((size_t)Bn * GH  * 2);   // [B,128]
  u16*   g2   = (u16*)  carve((size_t)Bn * GH  * 2);   // [B,128]
  float* bcb  = (float*)carve((size_t)Bn * En  * 4);   // [B,8]
  u16*   wbuf = (u16*)  carve((size_t)En * Hn * K12 * 2); // reused per layer (max W)
  u16*   hA   = (u16*)  carve((size_t)Bn * K12 * 2);   // [B,1088]
  u16*   hB   = (u16*)  carve((size_t)Bn * K12 * 2);   // [B,1088]
  u16*   hC   = (u16*)  carve((size_t)Bn * Hn  * 2);   // [B,1024]
  (void)ws_size; (void)in_sizes; (void)n_in; (void)out_size;

  const dim3 cvtB(256), cvtG(2048);

  // gate weights -> bf16; activations
  k_f32_to_bf16<<<cvtG, cvtB, 0, stream>>>(Wg1, wg1b, GH * K0);
  k_f32_to_bf16<<<cvtG, cvtB, 0, stream>>>(Wg2, wg2b, GH * GH);
  k_concat_xz<<<cvtG, cvtB, 0, stream>>>(x, z, gbuf);
  k_zfill<<<1024, 256, 0, stream>>>(z, hA, hB);

  // gate network
  k_gate_gemm<<<dim3(Bn / 64, GH / 16), dim3(32, 4), 0, stream>>>(gbuf, wg1b, bg1, g1, K0, GH);
  k_gate_gemm<<<dim3(Bn / 64, GH / 16), dim3(32, 4), 0, stream>>>(g1, wg2b, bg2, g2, GH, GH);
  k_gate3_softmax<<<Bn / 256, 256, 0, stream>>>(g2, Wg3, bg3, bcb);

  const dim3 blk(32, 4);
  // L0: [B,864] x W0[8,1024,864] -> hA[:, :1024], ELU
  k_f32_to_bf16<<<cvtG, cvtB, 0, stream>>>(W0, wbuf, En * Hn * K0);
  k_blended<true, false><<<dim3(Bn / 128, Hn / 64), blk, 0, stream>>>(
      gbuf, K0, wbuf, b0, bcb, hA, K12, K0, Hn);
  // L1: [B,1088] x W1[8,1024,1088] -> hB[:, :1024], ELU
  k_f32_to_bf16<<<cvtG, cvtB, 0, stream>>>(W1, wbuf, En * Hn * K12);
  k_blended<true, false><<<dim3(Bn / 128, Hn / 64), blk, 0, stream>>>(
      hA, K12, wbuf, b1, bcb, hB, K12, K12, Hn);
  // L2: [B,1088] x W2 -> hC[:, :1024], ELU
  k_f32_to_bf16<<<cvtG, cvtB, 0, stream>>>(W2, wbuf, En * Hn * K12);
  k_blended<true, false><<<dim3(Bn / 128, Hn / 64), blk, 0, stream>>>(
      hB, K12, wbuf, b2, bcb, hC, Hn, K12, Hn);
  // L3: [B,1024] x W3[8,768,1024] -> out f32, no activation
  k_f32_to_bf16<<<cvtG, cvtB, 0, stream>>>(W3, wbuf, En * ODn * Hn);
  k_blended<false, true><<<dim3(Bn / 128, ODn / 64), blk, 0, stream>>>(
      hC, Hn, wbuf, b3, bcb, out, ODn, Hn, ODn);
}
